// Ansatz_fb_82961588289571
// MI455X (gfx1250) — matrix operand order repaired and checked by reading, not compiled, not run on hardware
//
#include <hip/hip_runtime.h>
#include <hip/hip_bf16.h>

typedef __attribute__((ext_vector_type(16))) __bf16 v16bf;
typedef __attribute__((ext_vector_type(8)))  __bf16 v8bf;
typedef __attribute__((ext_vector_type(16))) float  v16f;
typedef __attribute__((ext_vector_type(8)))  float  v8f;

#define BATCH 1024
#define NE 32
#define NSV 256
#define NPV 32
#define CATW 832   // padded cat row width (bf16), 1664B row stride, 16B aligned

__device__ __forceinline__ v8f wmma_bf16(v16bf a, v16bf b, v8f c) {
  return __builtin_amdgcn_wmma_f32_16x16x32_bf16(false, a, false, b, (short)0, c,
                                                 false, false);
}

// A-fragment, contiguous k order: halves 0..15 -> k0..k0+15
// (k0 = kt*32 + grp*16, 16B aligned chunks).
__device__ __forceinline__ v16bf load_afrag(const __bf16* row_k0) {
  v8bf lo = *(const v8bf*)(row_k0);
  v8bf hi = *(const v8bf*)(row_k0 + 8);
  return __builtin_shufflevector(lo, hi, 0, 1, 2, 3, 4, 5, 6, 7,
                                         8, 9, 10, 11, 12, 13, 14, 15);
}

// B-fragment: 16 contiguous fp32 weights (64B aligned) -> bf16
__device__ __forceinline__ v16bf load_bfrag_f32(const float* p) {
  v16f w = *(const v16f*)p;
  return __builtin_convertvector(w, v16bf);
}

__global__ __launch_bounds__(256)
void ansatz_kernel(const float* __restrict__ r,   const float* __restrict__ a,
                   const float* __restrict__ V0w, const float* __restrict__ V0b,
                   const float* __restrict__ V1w, const float* __restrict__ V1b,
                   const float* __restrict__ V2w, const float* __restrict__ V2b,
                   const float* __restrict__ W0w, const float* __restrict__ W0b,
                   const float* __restrict__ W1w, const float* __restrict__ W1b,
                   const float* __restrict__ Vhuw,const float* __restrict__ Vhub,
                   const float* __restrict__ Vhdw,const float* __restrict__ Vhdb,
                   const float* __restrict__ wuw, const float* __restrict__ wub,
                   const float* __restrict__ wdw, const float* __restrict__ wdb,
                   float* __restrict__ out_ld, float* __restrict__ out_sg) {
  extern __shared__ unsigned char smem_raw[];
  float*  s_v    = (float*)smem_raw;          // 32*256
  float*  s_mean = s_v    + NE * NSV;         // 2*256  (su | sd)
  float*  p_mean = s_mean + 512;              // 2*32*32 (pu | pd)
  float*  orb    = p_mean + 2048;             // 2*16*16
  float*  r_loc  = orb    + 512;              // 32*3
  float*  a_loc  = r_loc  + 96;               // 4*3
  float*  esum   = a_loc  + 12;               // 32
  float*  dres   = esum   + 32;               // 4
  __bf16* catb   = (__bf16*)(dres + 4);       // 32*832 concat matrix (bf16)
  __bf16* pvb    = catb + NE * CATW;          // 32*32*32 pair features
  __bf16* sud    = pvb + NE * NE * NPV;       // 2*16*128 head activations

  const int b    = blockIdx.x;
  const int tid  = threadIdx.x;
  const int wave = tid >> 5;
  const int lane = tid & 31;
  const int grp  = lane >> 4;
  const int lm   = lane & 15;

  __builtin_prefetch(V1w + (tid << 6), 0, 0);
  __builtin_prefetch(V2w + (tid << 6), 0, 0);

  // ---- phase 0: features -------------------------------------------------
  if (tid < NE * 3) r_loc[tid] = r[b * NE * 3 + tid];
  if (tid < 12)     a_loc[tid] = a[tid];
  __syncthreads();

  if (tid < NE) {
    float ex = 0.f;
    #pragma unroll
    for (int at = 0; at < 4; ++at) {
      float dx = r_loc[tid * 3 + 0] - a_loc[at * 3 + 0];
      float dy = r_loc[tid * 3 + 1] - a_loc[at * 3 + 1];
      float dz = r_loc[tid * 3 + 2] - a_loc[at * 3 + 2];
      float len = sqrtf(dx * dx + dy * dy + dz * dz);
      s_v[tid * NSV + at * 4 + 0] = dx;
      s_v[tid * NSV + at * 4 + 1] = dy;
      s_v[tid * NSV + at * 4 + 2] = dz;
      s_v[tid * NSV + at * 4 + 3] = len;
      ex += expf(-len);
    }
    esum[tid] = ex;
  }
  for (int p = tid; p < NE * NE; p += 256) {   // rr[i][j] = r[j]-r[i]
    int i = p >> 5, j = p & 31;
    float dx = r_loc[j * 3 + 0] - r_loc[i * 3 + 0];
    float dy = r_loc[j * 3 + 1] - r_loc[i * 3 + 1];
    float dz = r_loc[j * 3 + 2] - r_loc[i * 3 + 2];
    float len = (i == j) ? 0.f : sqrtf(dx * dx + dy * dy + dz * dz);
    __bf16* row = pvb + p * NPV;
    row[0] = (__bf16)dx; row[1] = (__bf16)dy;
    row[2] = (__bf16)dz; row[3] = (__bf16)len;
    for (int f = 4; f < NPV; ++f) row[f] = (__bf16)0.f;  // zero-pad (A-frag safe)
  }
  __syncthreads();

  const float* Vw_arr[3] = {V0w, V1w, V2w};
  const float* Vb_arr[3] = {V0b, V1b, V2b};

  // ---- layer loop: fully unrolled so each layer is branch-specialized ----
  #pragma unroll
  for (int l = 0; l < 3; ++l) {
    const int suw    = (l == 0) ? 16 : 256;
    const int pw     = (l == 0) ? 4 : 32;
    const int suw_sh = (l == 0) ? 4 : 8;
    const int pw_sh  = (l == 0) ? 2 : 5;

    // means (su, sd, pu, pd)
    if (tid < suw) {
      float au = 0.f, ad = 0.f;
      for (int i = 0; i < 16; ++i) {
        au += s_v[i * NSV + tid];
        ad += s_v[(16 + i) * NSV + tid];
      }
      s_mean[tid]       = au * (1.f / 16.f);
      s_mean[256 + tid] = ad * (1.f / 16.f);
    }
    for (int idx = tid; idx < 2 * NE * NPV; idx += 256) {
      int side = idx >> 10, e = (idx >> 5) & 31, f = idx & 31;
      float acc = 0.f;
      int ib = side * 16;
      for (int i = 0; i < 16; ++i) acc += (float)pvb[((ib + i) * NE + e) * NPV + f];
      p_mean[idx] = acc * (1.f / 16.f);
    }
    __syncthreads();

    // build cat matrix in LDS (uniform, region-partitioned loops)
    {
      const int c2 = 2 * suw, c3 = 2 * suw + pw, c4 = 2 * suw + 2 * pw;
      for (int idx = tid; idx < NE * suw; idx += 256) {
        int e = idx >> suw_sh, k = idx & (suw - 1);
        __bf16* row = catb + e * CATW;
        row[k]        = (__bf16)s_mean[k];
        row[suw + k]  = (__bf16)s_mean[256 + k];
        row[c4 + k]   = (__bf16)s_v[e * NSV + k];
      }
      for (int idx = tid; idx < NE * pw; idx += 256) {
        int e = idx >> pw_sh, k = idx & (pw - 1);
        __bf16* row = catb + e * CATW;
        row[c2 + k] = (__bf16)p_mean[e * NPV + k];
        row[c3 + k] = (__bf16)p_mean[1024 + e * NPV + k];
      }
      if (l == 0) {  // zero-pad cols [56,64)
        for (int idx = tid; idx < NE * 8; idx += 256) {
          int e = idx >> 3, k = idx & 7;
          catb[e * CATW + 56 + k] = (__bf16)0.f;
        }
      }
    }

    // p update (layers 0,1): (1024 x pw) @ (pw x 32) via WMMA.
    // Same phase as cat-build (touches only pvb rows this wave owns).
    if (l < 2) {
      for (int s = 0; s < 8; ++s) {
        const int p0 = (wave + s * 8) * 16;
        v8f acc[2] = {};
        v16bf af = load_afrag(pvb + (p0 + lm) * NPV + grp * 16);
        #pragma unroll
        for (int t = 0; t < 2; ++t) {
          const int n = t * 16 + lm;
          v16bf bfv;
          if (l == 1) {
            bfv = load_bfrag_f32(W1w + n * 32 + grp * 16);
          } else {
            #pragma unroll
            for (int h = 0; h < 16; ++h) {  // branchless clamp+select, K=4
              int k = grp * 16 + h;
              float w = W0w[n * 4 + ((k < 4) ? k : 0)] * ((k < 4) ? 1.f : 0.f);
              bfv[h] = (__bf16)w;
            }
          }
          acc[t] = wmma_bf16(af, bfv, acc[t]);
        }
        const float* Wb = (l == 1) ? W1b : W0b;
        #pragma unroll
        for (int t = 0; t < 2; ++t) {
          #pragma unroll
          for (int vv = 0; vv < 8; ++vv) {
            int row = p0 + grp * 8 + vv;
            int col = t * 16 + lm;
            float old = (float)pvb[row * NPV + col];
            float val = tanhf(acc[t][vv] + Wb[col]);
            if (l == 1) val += old;
            pvb[row * NPV + col] = (__bf16)val;
          }
        }
      }
    }
    __syncthreads();

    // s update: (32 x Kcat) @ (Kcat x 256), writes s_v directly
    {
      const float* Vw = Vw_arr[l];
      const float* Vb = Vb_arr[l];
      const int mt = wave & 1;
      const int nb = (wave >> 1) * 4;
      v8f acc[4] = {};
      const int nkt = (l == 0) ? 2 : 26;
      for (int kt = 0; kt < nkt; ++kt) {
        v16bf af = load_afrag(catb + (mt * 16 + lm) * CATW + kt * 32 + grp * 16);
        #pragma unroll
        for (int t = 0; t < 4; ++t) {
          const int n = (nb + t) * 16 + lm;
          v16bf bfv;
          if (l != 0) {
            bfv = load_bfrag_f32(Vw + n * 832 + kt * 32 + grp * 16);
          } else {
            #pragma unroll
            for (int h = 0; h < 16; ++h) {  // branchless, K=56
              int k = kt * 32 + grp * 16 + h;
              float w = Vw[n * 56 + ((k < 56) ? k : 0)] * ((k < 56) ? 1.f : 0.f);
              bfv[h] = (__bf16)w;
            }
          }
          acc[t] = wmma_bf16(af, bfv, acc[t]);
        }
      }
      __syncthreads();   // cat reads of s_v complete -> safe to overwrite s_v
      #pragma unroll
      for (int t = 0; t < 4; ++t) {
        const int n0 = (nb + t) * 16 + lm;
        #pragma unroll
        for (int vv = 0; vv < 8; ++vv) {
          int row = mt * 16 + grp * 8 + vv;
          s_v[row * NSV + n0] = tanhf(acc[t][vv] + Vb[n0]);
        }
      }
    }
    __syncthreads();
  }

  // ---- heads: stage bf16 copy of s_v into catb, then WMMA ----------------
  __bf16* sbf = catb;   // 32*256 bf16, row stride 256 (512B, aligned)
  for (int idx = tid; idx < NE * NSV; idx += 256) sbf[idx] = (__bf16)s_v[idx];
  __syncthreads();

  #pragma unroll
  for (int side = 0; side < 2; ++side) {
    const float* Hw = side ? Vhdw : Vhuw;
    const float* Hb = side ? Vhdb : Vhub;
    v8f acc = {};
    for (int kt = 0; kt < 8; ++kt) {
      v16bf af = load_afrag(sbf + (side * 16 + lm) * 256 + kt * 32 + grp * 16);
      v16bf bfv = load_bfrag_f32(Hw + (wave * 16 + lm) * 256 + kt * 32 + grp * 16);
      acc = wmma_bf16(af, bfv, acc);
    }
    #pragma unroll
    for (int vv = 0; vv < 8; ++vv) {
      int e = grp * 8 + vv;
      int col = wave * 16 + lm;
      sud[(side * 16 + e) * 128 + col] = (__bf16)tanhf(acc[vv] + Hb[col]);
    }
  }
  __syncthreads();

  // ---- orbital matrices: (16x128)@(128x16) + envelope --------------------
  if (wave < 2) {
    const int side = wave;
    const float* Ow = side ? wdw : wuw;
    const float* Ob = side ? wdb : wub;
    v8f acc = {};
    for (int kt = 0; kt < 4; ++kt) {
      v16bf af = load_afrag(sud + (side * 16 + lm) * 128 + kt * 32 + grp * 16);
      v16bf bfv = load_bfrag_f32(Ow + lm * 128 + kt * 32 + grp * 16);
      acc = wmma_bf16(af, bfv, acc);
    }
    #pragma unroll
    for (int vv = 0; vv < 8; ++vv) {
      int e = grp * 8 + vv;
      float val = acc[vv] + Ob[lm];
      orb[side * 256 + e * 16 + lm] = val * esum[side * 16 + e];
    }
  }
  __syncthreads();

  // ---- slogdet via serial 16x16 partial-pivot LU -------------------------
  if (tid < 2) {
    float* M = orb + tid * 256;
    float sgn = 1.f, ld = 0.f;
    for (int k = 0; k < 16; ++k) {
      int piv = k;
      float mx = fabsf(M[k * 16 + k]);
      for (int rr2 = k + 1; rr2 < 16; ++rr2) {
        float v2 = fabsf(M[rr2 * 16 + k]);
        if (v2 > mx) { mx = v2; piv = rr2; }
      }
      if (piv != k) {
        for (int c = 0; c < 16; ++c) {
          float tsw = M[k * 16 + c];
          M[k * 16 + c] = M[piv * 16 + c];
          M[piv * 16 + c] = tsw;
        }
        sgn = -sgn;
      }
      float pvv = M[k * 16 + k];
      if (pvv < 0.f) sgn = -sgn;
      ld += logf(fabsf(pvv));
      float inv = 1.f / pvv;
      for (int rr2 = k + 1; rr2 < 16; ++rr2) {
        float f2 = M[rr2 * 16 + k] * inv;
        for (int c = k + 1; c < 16; ++c) M[rr2 * 16 + c] -= f2 * M[k * 16 + c];
      }
    }
    dres[tid] = sgn;
    dres[2 + tid] = ld;
  }
  __syncthreads();
  if (tid == 0) {
    out_sg[b] = dres[0] * dres[1];
    out_ld[b] = dres[2] + dres[3];
  }
}

// ---- batch reduction: log|sum sign*exp(ld - max)| + max --------------------
__global__ __launch_bounds__(1024)
void reduce_kernel(const float* __restrict__ ld, const float* __restrict__ sg,
                   float* __restrict__ out) {
  __shared__ float smax[1024];
  __shared__ float ssum[1024];
  const int t = threadIdx.x;
  smax[t] = ld[t];
  __syncthreads();
  for (int s = 512; s > 0; s >>= 1) {
    if (t < s) smax[t] = fmaxf(smax[t], smax[t + s]);
    __syncthreads();
  }
  const float mx = smax[0];
  __syncthreads();
  ssum[t] = sg[t] * expf(ld[t] - mx);
  __syncthreads();
  for (int s = 512; s > 0; s >>= 1) {
    if (t < s) ssum[t] += ssum[t + s];
    __syncthreads();
  }
  if (t == 0) out[0] = logf(fabsf(ssum[0])) + mx;
}

extern "C" void kernel_launch(void* const* d_in, const int* in_sizes, int n_in,
                              void* d_out, int out_size, void* d_ws, size_t ws_size,
                              hipStream_t stream) {
  (void)in_sizes; (void)n_in; (void)out_size; (void)ws_size;
  const float* r    = (const float*)d_in[0];
  const float* a    = (const float*)d_in[1];
  const float* V0w  = (const float*)d_in[2];
  const float* V0b  = (const float*)d_in[3];
  const float* V1w  = (const float*)d_in[4];
  const float* V1b  = (const float*)d_in[5];
  const float* V2w  = (const float*)d_in[6];
  const float* V2b  = (const float*)d_in[7];
  const float* W0w  = (const float*)d_in[8];
  const float* W0b  = (const float*)d_in[9];
  const float* W1w  = (const float*)d_in[10];
  const float* W1b  = (const float*)d_in[11];
  const float* Vhuw = (const float*)d_in[12];
  const float* Vhub = (const float*)d_in[13];
  const float* Vhdw = (const float*)d_in[14];
  const float* Vhdb = (const float*)d_in[15];
  const float* wuw  = (const float*)d_in[16];
  const float* wub  = (const float*)d_in[17];
  const float* wdw  = (const float*)d_in[18];
  const float* wdb  = (const float*)d_in[19];

  float* ld = (float*)d_ws;        // BATCH floats
  float* sg = ld + BATCH;          // BATCH floats

  // dynamic LDS: floats 11408*4 = 45632 B; bf16 (26624+32768+4096)*2 = 126976 B
  const size_t smem = 45632 + 126976;   // 172608 B
  ansatz_kernel<<<BATCH, 256, smem, stream>>>(
      r, a, V0w, V0b, V1w, V1b, V2w, V2b, W0w, W0b, W1w, W1b,
      Vhuw, Vhub, Vhdw, Vhdb, wuw, wub, wdw, wdb, ld, sg);
  reduce_kernel<<<1, 1024, 0, stream>>>(ld, sg, (float*)d_out);
}
